// GGCNLayer_40510131536310
// MI455X (gfx1250) — matrix operand order repaired and checked by reading, not compiled, not run on hardware
//
#include <hip/hip_runtime.h>

typedef __attribute__((ext_vector_type(16))) _Float16 v16h;
typedef __attribute__((ext_vector_type(8)))  _Float16 v8h;
typedef __attribute__((ext_vector_type(8)))  float    v8f;
typedef __attribute__((ext_vector_type(4)))  float    v4f;

#define BB 2
#define NN 256
#define DD 256
// derived
#define BN (BB*NN)            // 512
#define BND (BB*NN*DD)        // 131072

// ---------------------------------------------------------------------------
// Kernel 1: node-path GEMMs in f32.  2048 blocks: mat = bid>>9, row = bid&511.
// Ah gets bA + bC folded in (edge update adds bC once per element).
// ---------------------------------------------------------------------------
__global__ __launch_bounds__(256) void prep_node_kernel(
    const float* __restrict__ h,
    const float* __restrict__ WA, const float* __restrict__ bA,
    const float* __restrict__ WB, const float* __restrict__ bB,
    const float* __restrict__ WU, const float* __restrict__ bU,
    const float* __restrict__ WV, const float* __restrict__ bV,
    const float* __restrict__ bC,
    float* __restrict__ Ah, float* __restrict__ Bh,
    float* __restrict__ Uh, float* __restrict__ Vh)
{
    __shared__ float hrow[DD];
    const int bid = blockIdx.x;
    const int mat = bid >> 9;
    const int row = bid & 511;          // b*N + n
    const int d   = threadIdx.x;
    hrow[d] = h[row * DD + d];
    __syncthreads();

    const float* W; const float* bias; float* out; float extra = 0.0f;
    if (mat == 0)      { W = WA; bias = bA; out = Ah; extra = bC[d]; }
    else if (mat == 1) { W = WB; bias = bB; out = Bh; }
    else if (mat == 2) { W = WU; bias = bU; out = Uh; }
    else               { W = WV; bias = bV; out = Vh; }

    float acc = bias[d] + extra;
#pragma unroll 8
    for (int k = 0; k < DD; ++k)
        acc = fmaf(hrow[k], W[k * DD + d], acc);
    out[row * DD + d] = acc;
}

// ---------------------------------------------------------------------------
// Kernel 2: (a) WC -> f16 pre-swizzled into WMMA B-operand order
//   flat half index f = ((kk*16 + nt)*32 + lane)*16 + t
//   k = kk*32 + (lane<16 ? 0 : 8) + (t<8 ? t : t+8) ;  n = nt*16 + (lane&15)
// (b) invdeg[row] = 1 / sum_j adj[row][j]
// Grid: 256 blocks for (a) + 2 blocks for (b) = 258.
// ---------------------------------------------------------------------------
__global__ __launch_bounds__(256) void prep_misc_kernel(
    const float* __restrict__ WC, const int* __restrict__ adj,
    _Float16* __restrict__ WCh, float* __restrict__ invdeg)
{
    const int bid = blockIdx.x, tid = threadIdx.x;
    if (bid < 256) {
        const int f    = bid * 256 + tid;
        const int t    = f & 15;
        const int lane = (f >> 4) & 31;
        const int nt   = (f >> 9) & 15;
        const int kk   = f >> 13;
        const int k    = kk * 32 + ((lane < 16) ? 0 : 8) + ((t < 8) ? t : t + 8);
        const int n    = nt * 16 + (lane & 15);
        WCh[f] = (_Float16)WC[k * DD + n];
    } else {
        const int row = (bid - 256) * 256 + tid;   // 0..511
        const int* a = adj + row * NN;
        int s = 0;
        for (int j = 0; j < NN; ++j) s += a[j];
        invdeg[row] = 1.0f / (float)s;
    }
}

// ---------------------------------------------------------------------------
// Kernel 3: fused edge update + gating + partial message reduction.
// Block = (b,i, j-tile of 16).  8 waves; wave w owns d-out tiles {2w, 2w+1}.
// GEMM tile: A = e[b,i, j0..j0+16, :] (f16, LDS), B = WCh, C += via
// v_wmma_f32_16x16x32_f16, K-loop of 8.
// ---------------------------------------------------------------------------
__global__ __launch_bounds__(256) void edge_kernel(
    const float* __restrict__ e,
    const float* __restrict__ Ah,
    const float* __restrict__ Bh,
    const float* __restrict__ Vh,
    const _Float16* __restrict__ WCh,
    float* __restrict__ e_out,
    float* __restrict__ partials)
{
    // 16 rows x 256 K halves, padded row stride 264 halves (528B) for banks
    __shared__ __align__(32) _Float16 E16[16 * 264];

    const int bid = blockIdx.x;          // ((b*N + i)*16 + jt)
    const int jt  = bid & 15;
    const int bi  = bid >> 4;            // b*N + i  (0..511)
    const int j0  = jt * 16;
    const int bjb = bi & 256;            // b*N  (b = bi>>8)
    const int tid = threadIdx.x;

    // --- stage e tile -> f16 LDS ---
    {
        const int row = tid >> 4;        // 0..15 (j within tile)
        const int seg = tid & 15;        // 16-element K segment
        const float* src = e + ((size_t)bi * NN + (j0 + row)) * DD + seg * 16;
        _Float16* dst = &E16[row * 264 + seg * 16];
#pragma unroll
        for (int t = 0; t < 16; t += 4) {
            v4f v = *(const v4f*)(src + t);
            dst[t + 0] = (_Float16)v.x;
            dst[t + 1] = (_Float16)v.y;
            dst[t + 2] = (_Float16)v.z;
            dst[t + 3] = (_Float16)v.w;
        }
    }
    __syncthreads();

    const int wave   = tid >> 5;
    const int lane   = tid & 31;
    const int n      = lane & 15;        // output column within tile
    const int rowm   = lane & 15;        // A-matrix row M = lane%16
    const int hiHalf = lane >> 4;        // lanes 16-31 hold the +8 K offsets
    const _Float16* arow = &E16[rowm * 264 + hiHalf * 8];
    const v16h* Wv = (const v16h*)WCh;

    for (int nt2 = 0; nt2 < 2; ++nt2) {
        const int nt = wave * 2 + nt2;
        v8f c = {};
#pragma unroll
        for (int kk = 0; kk < 8; ++kk) {
            // A operand: lane<16 -> K in [kk*32+0,8)∪[kk*32+16,24)
            //            lane>=16 -> K in [kk*32+8,16)∪[kk*32+24,32)
            v8h alo = *(const v8h*)(arow + kk * 32);
            v8h ahi = *(const v8h*)(arow + kk * 32 + 16);
            v16h a = __builtin_shufflevector(alo, ahi,
                0,1,2,3,4,5,6,7,8,9,10,11,12,13,14,15);
            // B operand: pre-swizzled, 32B contiguous per lane
            v16h bm = Wv[kk * 512 + nt * 32 + lane];
            c = __builtin_amdgcn_wmma_f32_16x16x32_f16(
                    false, a, false, bm, (short)0, c, false, false);
        }

        // --- epilogue: e_new -> relu -> +e -> e_out; sigmoid-gate * Vh ---
        const int   d     = nt * 16 + n;
        const int   mbase = hiHalf ? 8 : 0;   // C/D layout: lanes 16-31 hold M=r+8
        const float bh    = Bh[bi * DD + d];
        float s = 0.0f;
#pragma unroll
        for (int r = 0; r < 8; ++r) {
            const int j = j0 + mbase + r;
            const float ah  = Ah[(bjb + j) * DD + d];   // includes bC
            const float en  = c[r] + ah + bh;
            const float rel = fmaxf(en, 0.0f);
            const size_t ei = ((size_t)bi * NN + j) * DD + d;
            const float eo  = e[ei] + rel;
            e_out[ei] = eo;
            const float g = 1.0f / (1.0f + __expf(-eo));
            s += g * Vh[(bjb + j) * DD + d];
        }
        // combine M=0..7 (lanes 0-15) with M=8..15 (lanes 16-31): deterministic
        const float s2 = __shfl_xor(s, 16, 32);
        if (lane < 16)
            partials[((size_t)bi * 16 + jt) * DD + d] = s + s2;
    }
}

// ---------------------------------------------------------------------------
// Kernel 4: h_out = h + relu(Uh + (sum_jt partials) * invdeg)
// ---------------------------------------------------------------------------
__global__ __launch_bounds__(256) void node_out_kernel(
    const float* __restrict__ h, const float* __restrict__ Uh,
    const float* __restrict__ partials, const float* __restrict__ invdeg,
    float* __restrict__ h_out)
{
    const int idx = blockIdx.x * 256 + threadIdx.x;   // < 131072
    const int row = idx >> 8;
    const int d   = idx & 255;
    float s = 0.0f;
#pragma unroll
    for (int jt = 0; jt < 16; ++jt)
        s += partials[((size_t)row * 16 + jt) * DD + d];
    const float val = Uh[idx] + s * invdeg[row];
    h_out[idx] = h[idx] + fmaxf(val, 0.0f);
}

// ---------------------------------------------------------------------------
extern "C" void kernel_launch(void* const* d_in, const int* in_sizes, int n_in,
                              void* d_out, int out_size, void* d_ws, size_t ws_size,
                              hipStream_t stream) {
    const float* h   = (const float*)d_in[0];
    const float* e   = (const float*)d_in[1];
    const int*   adj = (const int*)  d_in[2];
    const float* WA  = (const float*)d_in[3];
    const float* bA  = (const float*)d_in[4];
    const float* WB  = (const float*)d_in[5];
    const float* bB  = (const float*)d_in[6];
    const float* WC  = (const float*)d_in[7];
    const float* bC  = (const float*)d_in[8];
    const float* WU  = (const float*)d_in[9];
    const float* bU  = (const float*)d_in[10];
    const float* WV  = (const float*)d_in[11];
    const float* bV  = (const float*)d_in[12];

    float* ws = (float*)d_ws;
    float* Ah       = ws;                    // 131072
    float* Bh       = ws + 131072;           // 131072
    float* Uh       = ws + 262144;           // 131072
    float* Vh       = ws + 393216;           // 131072
    float* partials = ws + 524288;           // 512*16*256 = 2097152
    float* invdeg   = ws + 2621440;          // 512
    _Float16* WCh   = (_Float16*)(ws + 2621952); // 65536 halves

    float* h_out = (float*)d_out;            // [B,N,D]
    float* e_out = h_out + BND;              // [B,N,N,D]

    prep_node_kernel<<<2048, 256, 0, stream>>>(h, WA, bA, WB, bB, WU, bU,
                                               WV, bV, bC, Ah, Bh, Uh, Vh);
    prep_misc_kernel<<<258, 256, 0, stream>>>(WC, adj, WCh, invdeg);
    edge_kernel<<<BN * (NN / 16), 256, 0, stream>>>(e, Ah, Bh, Vh, WCh,
                                                    e_out, partials);
    node_out_kernel<<<BND / 256, 256, 0, stream>>>(h, Uh, partials, invdeg,
                                                   h_out);
}
